// Generator_8727373546102
// MI455X (gfx1250) — compile-verified, matrix operand order checked by reading, not run on hardware
//
#include <hip/hip_runtime.h>
#include <hip/hip_bf16.h>

// ---------------- shapes ----------------
#define EPSV 1e-5f
#define B_   8
#define CC   32
#define TT   504
#define FP   168     // padded F
#define FO   161     // original F
#define HH   256     // LSTM hidden
#define GG   1024    // 4*H gate dim
#define LC   21      // intra seq len  (FP/8)
#define LT   63      // inter seq len  (TT/8)
#define NI   4032    // B*TT   rows (intra LSTM batch)
#define N2   1344    // B*FP   rows (inter LSTM batch)
#define MF   84672   // NI*LC == N2*LT (fold GEMM rows)
#define NHD  4
#define EE   4
#define CVD  8
#define LOOK 5

typedef __bf16 bf16_t;
typedef __attribute__((ext_vector_type(16))) __bf16 v16bf;
typedef __attribute__((ext_vector_type(8)))  __bf16 bf16x8;
typedef __attribute__((ext_vector_type(8)))  float  v8f;

// ---------------- WMMA GEMM (bf16 in, f32 out) ----------------
// D[M,N] = concat_K(A0[M,K0], A1[M,K1]) * Bw[N,K]^T   (Bw row-major (N,K))
// A fragment 16x32 bf16 lane layout (CDNA5 ISA 7.12.2):
//   lane r(0..15)+16*half ; VGPR pairs hold K = (v<4?0:16) + 8*half + 2*(v%4)+p
// -> per lane: two contiguous 8-element (16B) loads at k+8*half and k+16+8*half.
__device__ __forceinline__ v16bf frag_ld(const bf16_t* p) {
  bf16x8 lo = *(const bf16x8*)(p);
  bf16x8 hi = *(const bf16x8*)(p + 16);
  v16bf r;
#pragma unroll
  for (int i = 0; i < 8; ++i) { r[i] = lo[i]; r[8 + i] = hi[i]; }
  return r;
}

__global__ void __launch_bounds__(128)
gemm_wmma_cat(const bf16_t* __restrict__ A0, long lda0, int K0,
              const bf16_t* __restrict__ A1, long lda1, int K1,
              const bf16_t* __restrict__ Bw, float* __restrict__ D, long ldd) {
  const int lane = threadIdx.x & 31;
  const int wave = threadIdx.x >> 5;
  const int r    = lane & 15;
  const int half = lane >> 4;
  const int m0   = blockIdx.x * 64 + wave * 16;
  const int n0   = blockIdx.y * 64;
  const int K    = K0 + K1;
  v8f acc0 = {}, acc1 = {}, acc2 = {}, acc3 = {};

  // per-lane base pointers (hoisted out of the K loops)
  const bf16_t* pa  = A0 + (long)(m0 + r) * lda0 + half * 8;
  const bf16_t* pb0 = Bw + (long)(n0 +  0 + r) * K + half * 8;
  const bf16_t* pb1 = Bw + (long)(n0 + 16 + r) * K + half * 8;
  const bf16_t* pb2 = Bw + (long)(n0 + 32 + r) * K + half * 8;
  const bf16_t* pb3 = Bw + (long)(n0 + 48 + r) * K + half * 8;

  // phase 1: K-columns from A0
  for (int k = 0; k < K0; k += 32) {
    v16bf a  = frag_ld(pa);
    v16bf b0 = frag_ld(pb0);
    v16bf b1 = frag_ld(pb1);
    v16bf b2 = frag_ld(pb2);
    v16bf b3 = frag_ld(pb3);
    __builtin_prefetch(pb0 + 32, 0, 3);   // gfx1250 global_prefetch_b8, next B K-slice
    acc0 = __builtin_amdgcn_wmma_f32_16x16x32_bf16(false, a, false, b0, (short)0, acc0, false, false);
    acc1 = __builtin_amdgcn_wmma_f32_16x16x32_bf16(false, a, false, b1, (short)0, acc1, false, false);
    acc2 = __builtin_amdgcn_wmma_f32_16x16x32_bf16(false, a, false, b2, (short)0, acc2, false, false);
    acc3 = __builtin_amdgcn_wmma_f32_16x16x32_bf16(false, a, false, b3, (short)0, acc3, false, false);
    pa += 32; pb0 += 32; pb1 += 32; pb2 += 32; pb3 += 32;
  }
  // phase 2: K-columns from A1 (K1 may be 0)
  const bf16_t* pc = A1 + (long)(m0 + r) * lda1 + half * 8;
  for (int k = 0; k < K1; k += 32) {
    v16bf a  = frag_ld(pc);
    v16bf b0 = frag_ld(pb0);
    v16bf b1 = frag_ld(pb1);
    v16bf b2 = frag_ld(pb2);
    v16bf b3 = frag_ld(pb3);
    __builtin_prefetch(pb0 + 32, 0, 3);
    acc0 = __builtin_amdgcn_wmma_f32_16x16x32_bf16(false, a, false, b0, (short)0, acc0, false, false);
    acc1 = __builtin_amdgcn_wmma_f32_16x16x32_bf16(false, a, false, b1, (short)0, acc1, false, false);
    acc2 = __builtin_amdgcn_wmma_f32_16x16x32_bf16(false, a, false, b2, (short)0, acc2, false, false);
    acc3 = __builtin_amdgcn_wmma_f32_16x16x32_bf16(false, a, false, b3, (short)0, acc3, false, false);
    pc += 32; pb0 += 32; pb1 += 32; pb2 += 32; pb3 += 32;
  }
  // C/D layout: VGPR i holds (M = i + 8*half, N = r)
#pragma unroll
  for (int i = 0; i < 8; ++i) {
    long row = (long)(m0 + i + 8 * half);
    D[row * ldd + n0 +  0 + r] = acc0[i];
    D[row * ldd + n0 + 16 + r] = acc1[i];
    D[row * ldd + n0 + 32 + r] = acc2[i];
    D[row * ldd + n0 + 48 + r] = acc3[i];
  }
}

// ---------------- small helpers ----------------
__global__ void zero_f32_k(float* p, long n) {
  long i = (long)blockIdx.x * 256 + threadIdx.x; if (i < n) p[i] = 0.f;
}
__global__ void zero_bf16_k(bf16_t* p, long n) {
  long i = (long)blockIdx.x * 256 + threadIdx.x; if (i < n) p[i] = (bf16_t)0.f;
}
// dst(1024 x 512) = [wih(1024x256) | whh(1024x256)]  -> bf16
__global__ void wcat_k(bf16_t* dst, const float* wih, const float* whh) {
  long i = (long)blockIdx.x * 256 + threadIdx.x;
  if (i >= (long)GG * 512) return;
  int k = i & 511; long g = i >> 9;
  float v = (k < HH) ? wih[g * HH + k] : whh[g * HH + (k - HH)];
  dst[i] = (bf16_t)v;
}
// dst[n*D+d] = src[d*N+n]  (transpose-convert fold weights)
__global__ void lint_k(bf16_t* dst, const float* src, int N, int D) {
  long i = (long)blockIdx.x * 256 + threadIdx.x;
  if (i >= (long)N * D) return;
  int d = i % D; int n = i / D;
  dst[i] = (bf16_t)src[(long)d * N + n];
}

// ---------------- front end ----------------
__global__ void conv_in_k(const float* x, const float* w, const float* bias, float* y) {
  long idx = (long)blockIdx.x * 256 + threadIdx.x;
  if (idx >= (long)B_ * CC * TT * FO) return;
  int f = idx % FO; long rr = idx / FO; int t = rr % TT; rr /= TT;
  int o = rr % CC; int b = rr / CC;
  float acc = bias[o];
  for (int ci = 0; ci < 2; ++ci)
    for (int kh = 0; kh < 3; ++kh) { int ti = t + kh - 1; if (ti < 0 || ti >= TT) continue;
      for (int kw = 0; kw < 3; ++kw) { int fi = f + kw - 1; if (fi < 0 || fi >= FO) continue;
        acc += x[(((long)b * 2 + ci) * TT + ti) * FO + fi] * w[((o * 2 + ci) * 3 + kh) * 3 + kw]; } }
  y[idx] = acc;
}
__global__ void gn_stats_k(const float* X, float* st) {
  int b = blockIdx.x; const long n = (long)CC * TT * FO;
  __shared__ float s1[256], s2[256];
  float a = 0.f, q = 0.f;
  for (long i = threadIdx.x; i < n; i += 256) {
    float v = X[(long)b * n + i]; a += v; q += v * v;
  }
  s1[threadIdx.x] = a; s2[threadIdx.x] = q; __syncthreads();
  for (int o = 128; o > 0; o >>= 1) {
    if (threadIdx.x < o) { s1[threadIdx.x] += s1[threadIdx.x + o]; s2[threadIdx.x] += s2[threadIdx.x + o]; }
    __syncthreads();
  }
  if (threadIdx.x == 0) { float mu = s1[0] / n; st[b * 2] = mu; st[b * 2 + 1] = s2[0] / n - mu * mu; }
}
__global__ void gn_apply_k(float* X, const float* g, const float* be, const float* st) {
  long idx = (long)blockIdx.x * 256 + threadIdx.x;
  if (idx >= (long)B_ * CC * TT * FO) return;
  int c = (idx / ((long)TT * FO)) % CC; int b = idx / ((long)CC * TT * FO);
  float inv = 1.f / sqrtf(st[b * 2 + 1] + EPSV);
  X[idx] = (X[idx] - st[b * 2]) * inv * g[c] + be[c];
}
__global__ void pad_k(const float* X, float* Xp) {
  long idx = (long)blockIdx.x * 256 + threadIdx.x;
  if (idx >= (long)B_ * CC * TT * FP) return;
  int f = idx % FP; long rr = idx / FP; int t = rr % TT; rr /= TT;
  int c = rr % CC; int b = rr / CC;
  Xp[idx] = (f < FO) ? X[(((long)b * CC + c) * TT + t) * FO + f] : 0.f;
}

// ---------------- LN over channels + unfold (writes bf16 LSTM input) ----------------
__global__ void lnc_unfold_intra_k(const float* xpad, const float* g, const float* be, bf16_t* z) {
  long idx = (long)blockIdx.x * 256 + threadIdx.x;
  if (idx >= (long)B_ * TT * FP) return;
  int f = idx % FP; long rr = idx / FP; int t = rr % TT; int b = rr / TT;
  const float* xp = xpad + ((long)b * CC * TT + t) * FP + f;
  float xs[CC]; float mu = 0.f;
#pragma unroll
  for (int c = 0; c < CC; ++c) { xs[c] = xp[(long)c * TT * FP]; mu += xs[c]; }
  mu *= (1.f / CC);
  float var = 0.f;
#pragma unroll
  for (int c = 0; c < CC; ++c) { float d = xs[c] - mu; var += d * d; }
  var *= (1.f / CC);
  float inv = 1.f / (sqrtf(var + EPSV) + EPSV);
  long n = (long)b * TT + t; int lblk = f >> 3; int kk = f & 7;
  bf16_t* zp = z + ((n * LC + lblk) << 8) + kk;
#pragma unroll
  for (int c = 0; c < CC; ++c) zp[c * 8] = (bf16_t)((xs[c] - mu) * inv * g[c] + be[c]);
}
__global__ void lnc_unfold_inter_k(const float* intra, const float* g, const float* be, bf16_t* z) {
  long idx = (long)blockIdx.x * 256 + threadIdx.x;
  if (idx >= (long)B_ * FP * TT) return;
  int t = idx % TT; long rr = idx / TT; int f = rr % FP; int b = rr / FP;
  const float* xp = intra + ((long)b * CC * TT + t) * FP + f;
  float xs[CC]; float mu = 0.f;
#pragma unroll
  for (int c = 0; c < CC; ++c) { xs[c] = xp[(long)c * TT * FP]; mu += xs[c]; }
  mu *= (1.f / CC);
  float var = 0.f;
#pragma unroll
  for (int c = 0; c < CC; ++c) { float d = xs[c] - mu; var += d * d; }
  var *= (1.f / CC);
  float inv = 1.f / (sqrtf(var + EPSV) + EPSV);
  long n2 = (long)b * FP + f; int lblk = t >> 3; int kk = t & 7;
  bf16_t* zp = z + ((n2 * LT + lblk) << 8) + kk;
#pragma unroll
  for (int c = 0; c < CC; ++c) zp[c * 8] = (bf16_t)((xs[c] - mu) * inv * g[c] + be[c]);
}

// ---------------- LSTM pointwise (gates -> c,h) ----------------
__global__ void lstm_pw_k(const float* gates, const float* bi, const float* bh,
                          float* cst, bf16_t* hdst, long hstride, int nrows) {
  long idx = (long)blockIdx.x * 256 + threadIdx.x;
  if (idx >= (long)nrows * HH) return;
  int j = idx % HH; long n = idx / HH;
  const float* gr = gates + n * GG;
  float gi = gr[j]          + bi[j]          + bh[j];
  float gf = gr[HH + j]     + bi[HH + j]     + bh[HH + j];
  float gg = gr[2 * HH + j] + bi[2 * HH + j] + bh[2 * HH + j];
  float go = gr[3 * HH + j] + bi[3 * HH + j] + bh[3 * HH + j];
  float si = 1.f / (1.f + __expf(-gi));
  float sf = 1.f / (1.f + __expf(-gf));
  float so = 1.f / (1.f + __expf(-go));
  float c  = sf * cst[idx] + si * tanhf(gg);
  cst[idx] = c;
  hdst[n * hstride + j] = (bf16_t)(so * tanhf(c));
}

// ---------------- fold pointwise (bias + residual, re-layout) ----------------
__global__ void fold_intra_pw_k(const float* y, const float* lb, const float* xpad, float* intra) {
  long idx = (long)blockIdx.x * 256 + threadIdx.x;
  if (idx >= (long)MF * 256) return;
  int q = idx & 255; long m = idx >> 8;
  int lblk = m % LC; long n = m / LC;
  int t = n % TT; int b = n / TT;
  int c = q >> 3; int f = lblk * 8 + (q & 7);
  long o = (((long)b * CC + c) * TT + t) * FP + f;
  intra[o] = y[idx] + lb[c] + xpad[o];
}
__global__ void fold_inter_pw_k(const float* y, const float* lb, const float* intra, float* outc) {
  long idx = (long)blockIdx.x * 256 + threadIdx.x;
  if (idx >= (long)MF * 256) return;
  int q = idx & 255; long m = idx >> 8;
  int lblk = m % LT; long n2 = m / LT;
  int f = n2 % FP; int b = n2 / FP;
  if (f >= FO) return;
  int c = q >> 3; int t = lblk * 8 + (q & 7);
  float v = y[idx] + lb[c] + intra[(((long)b * CC + c) * TT + t) * FP + f];
  outc[(((long)b * CC + c) * TT + t) * FO + f] = v;
}

// ---------------- QKV + per-head / proj layer norms + attention ----------------
__global__ void qkv_k(const float* X,
                      const float* qw, const float* qb, const float* qa,
                      const float* kw, const float* kb, const float* ka,
                      const float* vw, const float* vb, const float* va,
                      float* Q, float* Kb_, float* V) {
  long idx = (long)blockIdx.x * 256 + threadIdx.x;
  if (idx >= (long)B_ * TT * FO) return;
  int f = idx % FO; long rr = idx / FO; int t = rr % TT; int b = rr / TT;
  float xs[CC];
#pragma unroll
  for (int c = 0; c < CC; ++c) xs[c] = X[(((long)b * CC + c) * TT + t) * FO + f];
  for (int h = 0; h < NHD; ++h) {
    float aq = qa[h], ak = ka[h], av = va[h];
    for (int e = 0; e < EE; ++e) {
      float a1 = qb[h * EE + e], a2 = kb[h * EE + e];
#pragma unroll
      for (int c = 0; c < CC; ++c) {
        a1 += xs[c] * qw[(h * EE + e) * CC + c];
        a2 += xs[c] * kw[(h * EE + e) * CC + c];
      }
      a1 = (a1 >= 0.f) ? a1 : aq * a1;
      a2 = (a2 >= 0.f) ? a2 : ak * a2;
      long o = ((((long)(h * B_ + b)) * EE + e) * TT + t) * FO + f;
      Q[o] = a1; Kb_[o] = a2;
    }
    for (int cv = 0; cv < CVD; ++cv) {
      float a3 = vb[h * CVD + cv];
#pragma unroll
      for (int c = 0; c < CC; ++c) a3 += xs[c] * vw[(h * CVD + cv) * CC + c];
      a3 = (a3 >= 0.f) ? a3 : av * a3;
      V[((((long)(h * B_ + b)) * CVD + cv) * TT + t) * FO + f] = a3;
    }
  }
}
__global__ void lnhead_stats_k(const float* X, int Edim, float* st) {
  int bid = blockIdx.x;                  // (h*B + b)*T + t
  int t = bid % TT; int hb = bid / TT;
  const int n = Edim * FO;
  __shared__ float s1[128], s2[128];
  float a = 0.f, q = 0.f;
  for (int i = threadIdx.x; i < n; i += 128) {
    int e = i / FO, f = i % FO;
    float v = X[(((long)hb * Edim + e) * TT + t) * FO + f];
    a += v; q += v * v;
  }
  s1[threadIdx.x] = a; s2[threadIdx.x] = q; __syncthreads();
  for (int o = 64; o > 0; o >>= 1) {
    if (threadIdx.x < o) { s1[threadIdx.x] += s1[threadIdx.x + o]; s2[threadIdx.x] += s2[threadIdx.x + o]; }
    __syncthreads();
  }
  if (threadIdx.x == 0) { float mu = s1[0] / n; st[bid * 2] = mu; st[bid * 2 + 1] = s2[0] / n - mu * mu; }
}
__global__ void lnhead_apply_k(float* X, const float* g, const float* be, int Edim, const float* st) {
  long idx = (long)blockIdx.x * 256 + threadIdx.x;
  long total = (long)NHD * B_ * Edim * TT * FO;
  if (idx >= total) return;
  int f = idx % FO; long rr = idx / FO; int t = rr % TT; rr /= TT;
  int e = rr % Edim; rr /= Edim; int b = rr % B_; int h = rr / B_;
  int s = (h * B_ + b) * TT + t;
  float inv = 1.f / (sqrtf(st[s * 2 + 1] + EPSV) + EPSV);
  X[idx] = (X[idx] - st[s * 2]) * inv * g[(h * Edim + e) * FO + f] + be[(h * Edim + e) * FO + f];
}
__global__ void __launch_bounds__(64) attn_k(const float* Q, const float* Kb_, const float* V, float* O) {
  int bid = blockIdx.x;                 // (h*B + b)*T + t
  int t = bid % TT; int hb = bid / TT; int b = hb % B_; int h = hb / B_;
  int tid = threadIdx.x;
  __shared__ float sc[LOOK + 1];
  __shared__ float red[64];
  int s0 = t - LOOK; if (s0 < 0) s0 = 0;
  int ns = t - s0 + 1;
  const float scale = rsqrtf((float)(EE * FO));
  for (int si = 0; si < ns; ++si) {
    int s = s0 + si;
    float p = 0.f;
    for (int d = tid; d < EE * FO; d += 64) {
      int e = d / FO, f = d % FO;
      p += Q[((((long)hb) * EE + e) * TT + t) * FO + f] *
           Kb_[((((long)hb) * EE + e) * TT + s) * FO + f];
    }
    red[tid] = p; __syncthreads();
    for (int o = 32; o > 0; o >>= 1) { if (tid < o) red[tid] += red[tid + o]; __syncthreads(); }
    if (tid == 0) sc[si] = red[0] * scale;
    __syncthreads();
  }
  float w[LOOK + 1];
  float mx = -1e30f;
  for (int si = 0; si < ns; ++si) mx = fmaxf(mx, sc[si]);
  float sum = 0.f;
  for (int si = 0; si < ns; ++si) { w[si] = __expf(sc[si] - mx); sum += w[si]; }
  float inv = 1.f / sum;
  for (int si = 0; si < ns; ++si) w[si] *= inv;
  for (int d = tid; d < CVD * FO; d += 64) {
    int cv = d / FO, f = d % FO;
    float acc = 0.f;
    for (int si = 0; si < ns; ++si)
      acc += w[si] * V[((((long)hb) * CVD + cv) * TT + (s0 + si)) * FO + f];
    O[((((long)b) * 32 + h * CVD + cv) * TT + t) * FO + f] = acc;
  }
}
__global__ void proj_k(float* P, const float* w, const float* bias, const float* aP) {
  long idx = (long)blockIdx.x * 256 + threadIdx.x;
  if (idx >= (long)B_ * TT * FO) return;
  int f = idx % FO; long rr = idx / FO; int t = rr % TT; int b = rr / TT;
  float a = *aP;
  float os[CC], rs[CC];
#pragma unroll
  for (int c = 0; c < CC; ++c) os[c] = P[(((long)b * CC + c) * TT + t) * FO + f];
  for (int d = 0; d < CC; ++d) {
    float acc = bias[d];
#pragma unroll
    for (int c = 0; c < CC; ++c) acc += os[c] * w[d * CC + c];
    rs[d] = (acc >= 0.f) ? acc : a * acc;
  }
#pragma unroll
  for (int d = 0; d < CC; ++d) P[(((long)b * CC + d) * TT + t) * FO + f] = rs[d];
}
__global__ void lnproj_stats_k(const float* P, float* st) {
  int bid = blockIdx.x;                 // b*T + t
  int t = bid % TT; int b = bid / TT;
  const int n = CC * FO;
  __shared__ float s1[128], s2[128];
  float a = 0.f, q = 0.f;
  for (int i = threadIdx.x; i < n; i += 128) {
    int c = i / FO, f = i % FO;
    float v = P[(((long)b * CC + c) * TT + t) * FO + f];
    a += v; q += v * v;
  }
  s1[threadIdx.x] = a; s2[threadIdx.x] = q; __syncthreads();
  for (int o = 64; o > 0; o >>= 1) {
    if (threadIdx.x < o) { s1[threadIdx.x] += s1[threadIdx.x + o]; s2[threadIdx.x] += s2[threadIdx.x + o]; }
    __syncthreads();
  }
  if (threadIdx.x == 0) { float mu = s1[0] / n; st[bid * 2] = mu; st[bid * 2 + 1] = s2[0] / n - mu * mu; }
}
__global__ void lnproj_apply_k(const float* P, const float* g, const float* be,
                               const float* st, float* X) {
  long idx = (long)blockIdx.x * 256 + threadIdx.x;
  if (idx >= (long)B_ * CC * TT * FO) return;
  int f = idx % FO; long rr = idx / FO; int t = rr % TT; rr /= TT;
  int c = rr % CC; int b = rr / CC;
  int s = b * TT + t;
  float inv = 1.f / (sqrtf(st[s * 2 + 1] + EPSV) + EPSV);
  X[idx] = (P[idx] - st[s * 2]) * inv * g[c * FO + f] + be[c * FO + f] + X[idx];
}

// ---------------- back end ----------------
__global__ void deconv_k(const float* X, const float* w, const float* bias, float* y) {
  long idx = (long)blockIdx.x * 256 + threadIdx.x;
  if (idx >= (long)B_ * 2 * TT * FO) return;
  int f = idx % FO; long rr = idx / FO; int t = rr % TT; rr /= TT;
  int o = rr % 2; int b = rr / 2;
  float acc = bias[o];
  for (int c = 0; c < CC; ++c)
    for (int kh = 0; kh < 3; ++kh) { int ti = t + kh - 1; if (ti < 0 || ti >= TT) continue;
      for (int kw = 0; kw < 3; ++kw) { int fi = f + kw - 1; if (fi < 0 || fi >= FO) continue;
        // w2[o,c,kh,kw] = deconv_w[c,o,2-kh,2-kw]
        acc += X[(((long)b * CC + c) * TT + ti) * FO + fi] *
               w[((c * 2 + o) * 3 + (2 - kh)) * 3 + (2 - kw)]; } }
  y[idx] = acc;
}

// ---------------- host ----------------
namespace {
inline char* carve(char*& p, size_t bytes) {
  char* r = p; p += (bytes + 255) & ~(size_t)255; return r;
}
inline unsigned cdiv(long a, long b) { return (unsigned)((a + b - 1) / b); }
inline void launch_gemm(hipStream_t s, const bf16_t* A0, long lda0, int K0,
                        const bf16_t* A1, long lda1, int K1,
                        const bf16_t* Bw, float* D, long ldd, int M, int N) {
  dim3 g(M / 64, N / 64), blk(128);
  gemm_wmma_cat<<<g, blk, 0, s>>>(A0, lda0, K0, A1, lda1, K1, Bw, D, ldd);
}
} // namespace

extern "C" void kernel_launch(void* const* d_in, const int* in_sizes, int n_in,
                              void* d_out, int out_size, void* d_ws, size_t ws_size,
                              hipStream_t stream) {
  const float* x           = (const float*)d_in[0];
  const float* conv_w      = (const float*)d_in[1];
  const float* conv_b      = (const float*)d_in[2];
  const float* gn_g        = (const float*)d_in[3];
  const float* gn_b        = (const float*)d_in[4];
  const float* intra_ng    = (const float*)d_in[5];
  const float* intra_nb    = (const float*)d_in[6];
  const float* intra_wih   = (const float*)d_in[7];
  const float* intra_whh   = (const float*)d_in[8];
  const float* intra_bih   = (const float*)d_in[9];
  const float* intra_bhh   = (const float*)d_in[10];
  const float* intra_lw    = (const float*)d_in[11];
  const float* intra_lb    = (const float*)d_in[12];
  const float* inter_ng    = (const float*)d_in[13];
  const float* inter_nb    = (const float*)d_in[14];
  const float* inter_wih   = (const float*)d_in[15];
  const float* inter_whh   = (const float*)d_in[16];
  const float* inter_bih   = (const float*)d_in[17];
  const float* inter_bhh   = (const float*)d_in[18];
  const float* inter_lw    = (const float*)d_in[19];
  const float* inter_lb    = (const float*)d_in[20];
  const float* q_w = (const float*)d_in[21]; const float* q_b = (const float*)d_in[22];
  const float* q_a = (const float*)d_in[23]; const float* q_g = (const float*)d_in[24];
  const float* q_be = (const float*)d_in[25];
  const float* k_w = (const float*)d_in[26]; const float* k_b = (const float*)d_in[27];
  const float* k_a = (const float*)d_in[28]; const float* k_g = (const float*)d_in[29];
  const float* k_be = (const float*)d_in[30];
  const float* v_w = (const float*)d_in[31]; const float* v_b = (const float*)d_in[32];
  const float* v_a = (const float*)d_in[33]; const float* v_g = (const float*)d_in[34];
  const float* v_be = (const float*)d_in[35];
  const float* p_w = (const float*)d_in[36]; const float* p_b = (const float*)d_in[37];
  const float* p_a = (const float*)d_in[38]; const float* p_g = (const float*)d_in[39];
  const float* p_be = (const float*)d_in[40];
  const float* dec_w = (const float*)d_in[41];
  const float* dec_b = (const float*)d_in[42];

  char* wp = (char*)d_ws;
  float*  xcur  = (float*)carve(wp, sizeof(float) * (size_t)B_ * CC * TT * FO);
  float*  xpad  = (float*)carve(wp, sizeof(float) * (size_t)B_ * CC * TT * FP);
  float*  intra = (float*)carve(wp, sizeof(float) * (size_t)B_ * CC * TT * FP);
  float*  yfold = (float*)carve(wp, sizeof(float) * (size_t)MF * 256);
  float*  gates = (float*)carve(wp, sizeof(float) * (size_t)NI * GG);
  float*  cst   = (float*)carve(wp, sizeof(float) * (size_t)NI * HH);
  float*  stats = (float*)carve(wp, sizeof(float) * (size_t)NHD * B_ * TT * 2);
  float*  qbuf  = (float*)carve(wp, sizeof(float) * (size_t)NHD * B_ * EE * TT * FO);
  float*  kbuf  = (float*)carve(wp, sizeof(float) * (size_t)NHD * B_ * EE * TT * FO);
  float*  vbuf  = (float*)carve(wp, sizeof(float) * (size_t)NHD * B_ * CVD * TT * FO);
  float*  obuf  = (float*)carve(wp, sizeof(float) * (size_t)B_ * 32 * TT * FO);
  bf16_t* zbuf  = (bf16_t*)carve(wp, sizeof(bf16_t) * (size_t)MF * 256);
  bf16_t* hs    = (bf16_t*)carve(wp, sizeof(bf16_t) * (size_t)MF * 512);
  bf16_t* h0    = (bf16_t*)carve(wp, sizeof(bf16_t) * (size_t)NI * HH);
  bf16_t* wcat_f = (bf16_t*)carve(wp, sizeof(bf16_t) * (size_t)GG * 512);
  bf16_t* wcat_b = (bf16_t*)carve(wp, sizeof(bf16_t) * (size_t)GG * 512);
  bf16_t* wcat_i = (bf16_t*)carve(wp, sizeof(bf16_t) * (size_t)GG * 512);
  bf16_t* wlinTa = (bf16_t*)carve(wp, sizeof(bf16_t) * (size_t)256 * 512);
  bf16_t* wlinTe = (bf16_t*)carve(wp, sizeof(bf16_t) * (size_t)256 * 256);

  const long nBCTF = (long)B_ * CC * TT * FO;
  const long nBCTP = (long)B_ * CC * TT * FP;
  const long nBTF  = (long)B_ * TT * FO;

  // front end: conv3x3 -> global norm
  conv_in_k<<<cdiv(nBCTF, 256), 256, 0, stream>>>(x, conv_w, conv_b, xcur);
  gn_stats_k<<<B_, 256, 0, stream>>>(xcur, stats);
  gn_apply_k<<<cdiv(nBCTF, 256), 256, 0, stream>>>(xcur, gn_g, gn_b, stats);
  zero_bf16_k<<<cdiv((long)NI * HH, 256), 256, 0, stream>>>(h0, (long)NI * HH);

  for (int l = 0; l < 4; ++l) {
    // weight prep: fused [Wih|Whh] bf16, transposed fold weights
    wcat_k<<<cdiv((long)GG * 512, 256), 256, 0, stream>>>(
        wcat_f, intra_wih + (size_t)l * 2 * GG * HH, intra_whh + (size_t)l * 2 * GG * HH);
    wcat_k<<<cdiv((long)GG * 512, 256), 256, 0, stream>>>(
        wcat_b, intra_wih + (size_t)(l * 2 + 1) * GG * HH, intra_whh + (size_t)(l * 2 + 1) * GG * HH);
    wcat_k<<<cdiv((long)GG * 512, 256), 256, 0, stream>>>(
        wcat_i, inter_wih + (size_t)l * GG * HH, inter_whh + (size_t)l * GG * HH);
    lint_k<<<cdiv(256 * 512, 256), 256, 0, stream>>>(wlinTa, intra_lw + (size_t)l * 512 * 256, 256, 512);
    lint_k<<<cdiv(256 * 256, 256), 256, 0, stream>>>(wlinTe, inter_lw + (size_t)l * 256 * 256, 256, 256);

    // pad + intra LN/unfold
    pad_k<<<cdiv(nBCTP, 256), 256, 0, stream>>>(xcur, xpad);
    lnc_unfold_intra_k<<<cdiv((long)B_ * TT * FP, 256), 256, 0, stream>>>(
        xpad, intra_ng + l * CC, intra_nb + l * CC, zbuf);

    // intra BiLSTM: fused [z_t, h_{t-1}] GEMM (K=512) per step
    for (int dir = 0; dir < 2; ++dir) {
      const bf16_t* wc = dir ? wcat_b : wcat_f;
      const float* bi = intra_bih + (size_t)(l * 2 + dir) * GG;
      const float* bh = intra_bhh + (size_t)(l * 2 + dir) * GG;
      int coloff = dir ? HH : 0;
      zero_f32_k<<<cdiv((long)NI * HH, 256), 256, 0, stream>>>(cst, (long)NI * HH);
      for (int s2 = 0; s2 < LC; ++s2) {
        int t = dir ? (LC - 1 - s2) : s2;
        const bf16_t* A1; long lda1;
        if (s2 == 0) { A1 = h0; lda1 = HH; }
        else { int tp = dir ? (t + 1) : (t - 1); A1 = hs + (size_t)tp * 512 + coloff; lda1 = (long)LC * 512; }
        launch_gemm(stream, zbuf + (size_t)t * 256, (long)LC * 256, 256,
                    A1, lda1, 256, wc, gates, GG, NI, GG);
        lstm_pw_k<<<cdiv((long)NI * HH, 256), 256, 0, stream>>>(
            gates, bi, bh, cst, hs + (size_t)t * 512 + coloff, (long)LC * 512, NI);
      }
    }
    // intra fold: (MF,512)x(512,256) WMMA GEMM, then bias+residual re-layout
    launch_gemm(stream, hs, 512, 512, hs, 512, 0, wlinTa, yfold, 256, MF, 256);
    fold_intra_pw_k<<<cdiv((long)MF * 256, 256), 256, 0, stream>>>(
        yfold, intra_lb + l * CC, xpad, intra);

    // inter LN/unfold + LSTM
    lnc_unfold_inter_k<<<cdiv((long)B_ * FP * TT, 256), 256, 0, stream>>>(
        intra, inter_ng + l * CC, inter_nb + l * CC, zbuf);
    zero_f32_k<<<cdiv((long)N2 * HH, 256), 256, 0, stream>>>(cst, (long)N2 * HH);
    for (int s2 = 0; s2 < LT; ++s2) {
      const bf16_t* A1; long lda1;
      if (s2 == 0) { A1 = h0; lda1 = HH; }
      else { A1 = hs + (size_t)(s2 - 1) * 256; lda1 = (long)LT * 256; }
      launch_gemm(stream, zbuf + (size_t)s2 * 256, (long)LT * 256, 256,
                  A1, lda1, 256, wcat_i, gates, GG, N2, GG);
      lstm_pw_k<<<cdiv((long)N2 * HH, 256), 256, 0, stream>>>(
          gates, inter_bih + (size_t)l * GG, inter_bhh + (size_t)l * GG,
          cst, hs + (size_t)s2 * 256, (long)LT * 256, N2);
    }
    launch_gemm(stream, hs, 256, 256, hs, 256, 0, wlinTe, yfold, 256, MF, 256);
    fold_inter_pw_k<<<cdiv((long)MF * 256, 256), 256, 0, stream>>>(
        yfold, inter_lb + l * CC, intra, xcur);      // xcur := cropped "inter"

    // attention: QKV -> head LN -> banded softmax attention -> proj -> proj LN -> residual
    qkv_k<<<cdiv(nBTF, 256), 256, 0, stream>>>(
        xcur,
        q_w + (size_t)l * NHD * EE * CC, q_b + (size_t)l * NHD * EE, q_a + (size_t)l * NHD,
        k_w + (size_t)l * NHD * EE * CC, k_b + (size_t)l * NHD * EE, k_a + (size_t)l * NHD,
        v_w + (size_t)l * NHD * CVD * CC, v_b + (size_t)l * NHD * CVD, v_a + (size_t)l * NHD,
        qbuf, kbuf, vbuf);
    lnhead_stats_k<<<NHD * B_ * TT, 128, 0, stream>>>(qbuf, EE, stats);
    lnhead_apply_k<<<cdiv((long)NHD * B_ * EE * TT * FO, 256), 256, 0, stream>>>(
        qbuf, q_g + (size_t)l * NHD * EE * FO, q_be + (size_t)l * NHD * EE * FO, EE, stats);
    lnhead_stats_k<<<NHD * B_ * TT, 128, 0, stream>>>(kbuf, EE, stats);
    lnhead_apply_k<<<cdiv((long)NHD * B_ * EE * TT * FO, 256), 256, 0, stream>>>(
        kbuf, k_g + (size_t)l * NHD * EE * FO, k_be + (size_t)l * NHD * EE * FO, EE, stats);
    lnhead_stats_k<<<NHD * B_ * TT, 128, 0, stream>>>(vbuf, CVD, stats);
    lnhead_apply_k<<<cdiv((long)NHD * B_ * CVD * TT * FO, 256), 256, 0, stream>>>(
        vbuf, v_g + (size_t)l * NHD * CVD * FO, v_be + (size_t)l * NHD * CVD * FO, CVD, stats);
    attn_k<<<NHD * B_ * TT, 64, 0, stream>>>(qbuf, kbuf, vbuf, obuf);
    proj_k<<<cdiv(nBTF, 256), 256, 0, stream>>>(
        obuf, p_w + (size_t)l * CC * CC, p_b + (size_t)l * CC, p_a + l);
    lnproj_stats_k<<<B_ * TT, 128, 0, stream>>>(obuf, stats);
    lnproj_apply_k<<<cdiv(nBCTF, 256), 256, 0, stream>>>(
        obuf, p_g + (size_t)l * CC * FO, p_be + (size_t)l * CC * FO, stats, xcur);
  }

  // back end: transposed-conv equivalent (flipped-kernel conv)
  deconv_k<<<cdiv((long)B_ * 2 * TT * FO, 256), 256, 0, stream>>>(
      xcur, dec_w, dec_b, (float*)d_out);
}
